// EncoderSRU_46471546143397
// MI455X (gfx1250) — matrix operand order repaired and checked by reading, not compiled, not run on hardware
//
#include <hip/hip_runtime.h>
#include <hip/hip_bf16.h>

// ---------------------------------------------------------------------------
// EncoderSRU for MI455X (gfx1250, wave32, WMMA)
//   L=1024, B=16, D_IN=1024, n=512 (2n=1024), N_LAYERS=4, 6n=3072
// Pipeline per layer:
//   x(f32) -> bf16 ; u = x_bf16 @ W_bf16 (v_wmma_f32_16x16x32_bf16) ;
//   bidirectional SRU scan (sigmoid/tanh) -> h, c_final
// ---------------------------------------------------------------------------

typedef __attribute__((ext_vector_type(16))) __bf16 v16bf;
typedef __attribute__((ext_vector_type(8)))  __bf16 v8bf;
typedef __attribute__((ext_vector_type(4)))  __bf16 v4bf;
typedef __attribute__((ext_vector_type(8)))  float  v8f;
typedef __attribute__((ext_vector_type(4)))  float  v4f;

#define LSEQ   1024
#define BATCH  16
#define DK     1024      // K dim (input width of every layer)
#define DN     3072      // 6n
#define NHID   512       // n
#define MROWS  (LSEQ*BATCH)   // 16384
#define NTILES (DN/16)        // 192
#define KTILES (DK/32)        // 32

// -------------------- embedding gather: x0 = emb[tok] ----------------------
__global__ __launch_bounds__(256) void sru_embed(const int* __restrict__ tok,
                                                 const float* __restrict__ emb,
                                                 float* __restrict__ X) {
  int idx = blockIdx.x * blockDim.x + threadIdx.x;   // MROWS*256 threads
  int lb = idx >> 8;
  int k4 = (idx & 255) << 2;
  int t = tok[lb];
  *(v4f*)(X + (size_t)lb * DK + k4) =
      *(const v4f*)(emb + (size_t)t * DK + k4);
}

// ----------- repack W (f32, KxN row-major) into WMMA B fragments -----------
// Layout: Wp[layer][ntile][ktile][lane][16 bf16]
//   lane<16 : element i = W[kt*32 +     (i<8 ? i : i+8)][nt*16 + lane]
//   lane>=16: element i = W[kt*32 + 8 + (i<8 ? i : i+8)][nt*16 + lane-16]
__global__ __launch_bounds__(256) void sru_pack_w(const float* __restrict__ Ws,
                                                  __bf16* __restrict__ Wp) {
  int idx = blockIdx.x * blockDim.x + threadIdx.x;   // 4*192*32*32 threads
  int lane  = idx & 31;
  int kt    = (idx >> 5) & (KTILES - 1);
  int nt    = (idx >> 10) % NTILES;
  int layer = idx / (NTILES * KTILES * 32);
  if (layer >= 4) return;
  int col   = nt * 16 + (lane & 15);
  int kbase = kt * 32 + ((lane < 16) ? 0 : 8);
  const float* src = Ws + ((size_t)layer * DK + kbase) * DN + col;
  __bf16* dst = Wp + (((size_t)layer * NTILES * KTILES + (size_t)nt * KTILES + kt) * 32 + lane) * 16;
#pragma unroll
  for (int i = 0; i < 16; ++i) dst[i] = (__bf16)src[(size_t)(i < 8 ? i : i + 8) * DN];
}

// ------------------------- f32 -> bf16 conversion --------------------------
__global__ __launch_bounds__(256) void sru_cvt_bf16(const float* __restrict__ in,
                                                    __bf16* __restrict__ out) {
  size_t i = ((size_t)blockIdx.x * blockDim.x + threadIdx.x) * 4;
  v4f v = *(const v4f*)(in + i);
  v4bf o;
  o[0] = (__bf16)v[0]; o[1] = (__bf16)v[1];
  o[2] = (__bf16)v[2]; o[3] = (__bf16)v[3];
  *(v4bf*)(out + i) = o;
}

// -------------------------- WMMA GEMM: U = Xb @ W --------------------------
// M=16384, K=1024, N=3072. Block tile 128x128, 8 waves (4Mx2N), each wave
// computes a 32x64 subtile = 2x4 grid of 16x16 f32 accumulators.
__global__ __launch_bounds__(256) void sru_gemm(const __bf16* __restrict__ Xb,
                                                const __bf16* __restrict__ Wp,
                                                float* __restrict__ U) {
  const int lane = threadIdx.x & 31;
  const int w    = threadIdx.x >> 5;
  const int wm   = w & 3;         // 0..3  (M direction)
  const int wn   = w >> 2;        // 0..1  (N direction)
  const int m0   = blockIdx.y * 128 + wm * 32;
  const int n0   = blockIdx.x * 128 + wn * 64;

  v8f acc[2][4];
#pragma unroll
  for (int im = 0; im < 2; ++im)
#pragma unroll
    for (int jn = 0; jn < 4; ++jn)
#pragma unroll
      for (int g = 0; g < 8; ++g) acc[im][jn][g] = 0.0f;

  // A fragment addressing (16-bit A 16x32 ISA layout):
  //   lane<16: row = m0+im*16+lane,    K = kb+{0..7}  and kb+16+{0..7}
  //   lane>=16: row = m0+im*16+lane-16, K = kb+8+{0..7} and kb+24+{0..7}
  const int rowL  = lane & 15;
  const int koffA = (lane < 16) ? 0 : 8;
  const __bf16* aBase0 = Xb + (size_t)(m0 + rowL) * DK + koffA;
  const __bf16* aBase1 = Xb + (size_t)(m0 + 16 + rowL) * DK + koffA;

  // B fragments: nt = n0/16 + jn, contiguous 16 bf16 per lane per ktile
  const __bf16* bBase = Wp + (((size_t)(n0 >> 4) * KTILES) * 32 + lane) * 16;
  const size_t bNtStride = (size_t)KTILES * 32 * 16;   // elements per ntile
  const size_t bKtStride = (size_t)32 * 16;            // elements per ktile

  for (int kb = 0; kb < DK; kb += 32) {
    const int kt = kb >> 5;
    union { v16bf v; v8bf h[2]; } a[2], b[4];
    a[0].h[0] = *(const v8bf*)(aBase0 + kb);
    a[0].h[1] = *(const v8bf*)(aBase0 + kb + 16);
    a[1].h[0] = *(const v8bf*)(aBase1 + kb);
    a[1].h[1] = *(const v8bf*)(aBase1 + kb + 16);
#pragma unroll
    for (int jn = 0; jn < 4; ++jn)
      b[jn].v = *(const v16bf*)(bBase + (size_t)jn * bNtStride + (size_t)kt * bKtStride);

    // stream-ahead hints (global_prefetch_b8)
    __builtin_prefetch(aBase0 + kb + 128, 0, 0);
    __builtin_prefetch(aBase1 + kb + 128, 0, 0);
    __builtin_prefetch(bBase + (size_t)(kt + 2) * bKtStride, 0, 0);

#pragma unroll
    for (int im = 0; im < 2; ++im)
#pragma unroll
      for (int jn = 0; jn < 4; ++jn)
        acc[im][jn] = __builtin_amdgcn_wmma_f32_16x16x32_bf16(
            false, a[im].v, false, b[jn].v, (short)0, acc[im][jn], false, false);
  }

  // C/D f32 16x16 layout: VGPR g -> row g (lanes 0-15) / g+8 (lanes 16-31),
  // col = lane%16
  const int rowAdd = (lane < 16) ? 0 : 8;
  const int col    = lane & 15;
#pragma unroll
  for (int im = 0; im < 2; ++im)
#pragma unroll
    for (int jn = 0; jn < 4; ++jn)
#pragma unroll
      for (int g = 0; g < 8; ++g)
        U[(size_t)(m0 + im * 16 + rowAdd + g) * DN + (n0 + jn * 16 + col)] =
            acc[im][jn][g];
}

// --------------------- bidirectional SRU recurrence ------------------------
// one thread per (dir, batch, channel); 2*16*512 = 16384 threads
__global__ __launch_bounds__(256) void sru_scan(const float* __restrict__ U,
                                                const float* __restrict__ X,
                                                const float* __restrict__ bias,
                                                float* __restrict__ H,
                                                float* __restrict__ Cout) {
  int tid = blockIdx.x * blockDim.x + threadIdx.x;
  int j = tid & (NHID - 1);
  int b = (tid >> 9) & (BATCH - 1);
  int d = tid >> 13;                       // direction 0/1
  float bfv = bias[d * NHID + j];
  float brv = bias[2 * NHID + d * NHID + j];
  const size_t ubase = (size_t)d * (3 * NHID) + j;
  const size_t xbase = (size_t)d * NHID + j;
  int t = d ? (LSEQ - 1) : 0;
  const int dt = d ? -1 : 1;
  float c = 0.0f;
  for (int s = 0; s < LSEQ; ++s, t += dt) {
    const size_t rb = (size_t)t * BATCH + b;
    const float* up = U + rb * DN + ubase;
    float xt  = up[0];
    float fin = up[NHID];
    float rin = up[2 * NHID];
    float xp  = X[rb * DK + xbase];
    float f = 1.0f / (1.0f + __expf(-(fin + bfv)));
    float r = 1.0f / (1.0f + __expf(-(rin + brv)));
    c = f * c + (1.0f - f) * xt;
    float h = r * tanhf(c) + (1.0f - r) * xp;
    H[rb * DK + xbase] = h;
  }
  Cout[(size_t)b * (2 * NHID) + d * NHID + j] = c;
}

// ---------------------------------------------------------------------------
extern "C" void kernel_launch(void* const* d_in, const int* in_sizes, int n_in,
                              void* d_out, int out_size, void* d_ws, size_t ws_size,
                              hipStream_t stream) {
  const int*   tok = (const int*)d_in[0];
  // d_in[1] = input_lengths (unused by the reference computation)
  const float* emb = (const float*)d_in[2];
  const float* Ws  = (const float*)d_in[3];
  const float* bs  = (const float*)d_in[4];
  float* out = (float*)d_out;

  // workspace layout (bytes)
  char* ws = (char*)d_ws;
  float*  xA = (float*)(ws);                               //  64 MB
  float*  xB = (float*)(ws + ((size_t)64  << 20));         //  64 MB
  __bf16* xh = (__bf16*)(ws + ((size_t)128 << 20));        //  32 MB
  float*  U  = (float*)(ws + ((size_t)160 << 20));         // 192 MB
  __bf16* Wp = (__bf16*)(ws + ((size_t)352 << 20));        //  24 MB

  // 1) embedding gather
  sru_embed<<<MROWS * 256 / 256, 256, 0, stream>>>(tok, emb, xA);

  // 2) repack all 4 layers' weights into WMMA B-fragment order (bf16)
  sru_pack_w<<<(4 * NTILES * KTILES * 32) / 256, 256, 0, stream>>>(Ws, Wp);

  float* hiddenBase = out + (size_t)MROWS * DK;   // after final x

  float* cur = xA;
  float* nxt = xB;
  for (int l = 0; l < 4; ++l) {
    // 3a) convert layer input to bf16
    sru_cvt_bf16<<<(MROWS * DK / 4) / 256, 256, 0, stream>>>(cur, xh);

    // 3b) U = x_bf16 @ W_bf16   (v_wmma_f32_16x16x32_bf16)
    dim3 gg(DN / 128, MROWS / 128, 1);
    const __bf16* WpL = Wp + (size_t)l * NTILES * KTILES * 32 * 16;
    sru_gemm<<<gg, 256, 0, stream>>>(xh, WpL, U);

    // 3c) recurrence; last layer writes h straight into d_out
    float* H = (l < 3) ? nxt : out;
    float* C = hiddenBase + (size_t)l * BATCH * DK;
    const float* biasL = bs + (size_t)l * 4 * NHID;
    sru_scan<<<(2 * BATCH * NHID) / 256, 256, 0, stream>>>(U, cur, biasL, H, C);

    float* tmp = cur; cur = nxt; nxt = tmp;
  }
}
